// VNNBlockTwo_65034394796047
// MI455X (gfx1250) — matrix operand order repaired and checked by reading, not compiled, not run on hardware
//
#include <hip/hip_runtime.h>

// ---------------------------------------------------------------------------
// MI455X (gfx1250) implementation.
//
// Phase A: hypernet -> 126 (3x3 kernel + bias) sets          (1 block, tiny)
// Phase B: sequential conv recurrences (123 / 3 steps)        (2 blocks, 1024t)
// Phase C: pack W into WMMA B-fragment layout as bf16 hi/lo   (257 blocks)
// Phase D: x[131072,256] @ W[256,256] + b  via split-bf16 WMMA (grid 2048)
//          3x v_wmma_f32_16x16x32_bf16 per K=32 slice, truncating hi/lo split
//          of x done in-register (x is read exactly once from HBM; W is
//          L2-resident).  ~268 MB HBM traffic -> ~11.5 us roofline.
// ---------------------------------------------------------------------------

typedef __bf16          v16bf  __attribute__((ext_vector_type(16)));
typedef float           v8f    __attribute__((ext_vector_type(8)));
typedef unsigned short  u16x16 __attribute__((ext_vector_type(16)));
typedef unsigned int    v8u    __attribute__((ext_vector_type(8)));

__device__ __forceinline__ unsigned short f2bf_rne(float f) {
    unsigned u = __float_as_uint(f);
    u += 0x7FFFu + ((u >> 16) & 1u);          // round-to-nearest-even
    return (unsigned short)(u >> 16);
}
__device__ __forceinline__ float bf2f(unsigned short h) {
    return __uint_as_float(((unsigned)h) << 16);
}

// Cheap truncating hi/lo split: hi = top 16 bits, lo = trunc16(f - hi).
// f - hi is exact in fp32; combined hi+lo carries ~16 mantissa bits.
__device__ __forceinline__ void split_trunc(float f, unsigned short& hi, unsigned short& lo) {
    unsigned u = __float_as_uint(f);
    hi = (unsigned short)(u >> 16);
    float r = f - __uint_as_float(u & 0xFFFF0000u);
    lo = (unsigned short)(__float_as_uint(r) >> 16);
}

// ---------------------------------------------------------------------------
// Phase A: hypernet.  out[t*10+0..8] = 3x3 kernel (row major), out[t*10+9] = bias
// ---------------------------------------------------------------------------
__global__ void hyper_kernel(const float* __restrict__ wW1, const float* __restrict__ wb1,
                             const float* __restrict__ wW2, const float* __restrict__ wb2,
                             const float* __restrict__ bW1, const float* __restrict__ bb1,
                             const float* __restrict__ bW2, const float* __restrict__ bb2,
                             float* __restrict__ wk, float* __restrict__ bk)
{
    __shared__ float row[64];
    __shared__ float h[48];
    const int tid = threadIdx.x;            // 64 threads
    const float NEGL = -9.210340371976184f / 64.0f;   // -ln(10000)/d_model

    for (int rep = 0; rep < 2; ++rep) {
        const float* W1 = rep ? bW1 : wW1;
        const float* b1 = rep ? bb1 : wb1;
        const float* W2 = rep ? bW2 : wW2;
        const float* b2 = rep ? bb2 : wb2;
        float* out      = rep ? bk  : wk;
        const int steps = rep ? 3   : 123;

        for (int t = 0; t < steps; ++t) {
            {   // sinusoidal positional-encoding row `t`
                int i = tid >> 1;
                float dt  = expf((float)(2 * i) * NEGL);
                float ang = (float)t * dt;
                row[tid] = (tid & 1) ? cosf(ang) : sinf(ang);
            }
            __syncthreads();
            if (tid < 48) {
                float s = b1[tid];
                for (int k = 0; k < 64; ++k) s += row[k] * W1[k * 48 + tid];
                h[tid] = tanhf(s);
            }
            __syncthreads();
            if (tid < 10) {
                float s = b2[tid];
                for (int j = 0; j < 48; ++j) s += h[j] * W2[j * 10 + tid];
                out[t * 10 + tid] = s;
            }
            __syncthreads();
        }
    }
}

// ---------------------------------------------------------------------------
// Phase B: sequential pad(2)+3x3 valid conv recurrence.
// block 0: weight pyramid (123 steps, 10->256).  block 1: bias pyramid (3 steps).
// Single workgroup per recurrence: inter-step sync via s_barrier (cheapest
// possible -- the recurrence is latency bound, not throughput bound).
// Ping-pong buffers sit in global memory (<=262KB, hot in WGP$/L2).
// ---------------------------------------------------------------------------
__global__ __launch_bounds__(1024)
void conv_rec_kernel(const float* __restrict__ winit, const float* __restrict__ wk,
                     float* wA, float* wB,
                     const float* __restrict__ binit, const float* __restrict__ bk,
                     float* bA, float* bB)
{
    const float* init; const float* kb; float* A; float* B; int steps;
    if (blockIdx.x == 0) { init = winit; kb = wk; A = wA; B = wB; steps = 123; }
    else                 { init = binit; kb = bk; A = bA; B = bB; steps = 3;   }

    const int tid = threadIdx.x, nt = blockDim.x;
    for (int p = tid; p < 100; p += nt) A[p] = init[p];

    __shared__ float ks[10];
    int s = 10;
    for (int t = 0; t < steps; ++t) {
        __threadfence_block();
        __syncthreads();                       // prev step's stores visible
        if (tid < 10) ks[tid] = kb[t * 10 + tid];
        __syncthreads();
        const float k0 = ks[0], k1 = ks[1], k2 = ks[2],
                    k3 = ks[3], k4 = ks[4], k5 = ks[5],
                    k6 = ks[6], k7 = ks[7], k8 = ks[8], kb9 = ks[9];

        const float* in  = (t & 1) ? B : A;
        float*       out = (t & 1) ? A : B;
        const int o   = s + 2;
        const int qpr = (o + 3) >> 2;          // quads per row
        const int nq  = o * qpr;

        for (int q = tid; q < nq; q += nt) {
            int y  = q / qpr;
            int x0 = (q - y * qpr) << 2;
            // out(y,x) = sum_{dy,dx in [0,3)} k[dy][dx] * in[y+dy-2][x+dx-2]
            float r0[6], r1[6], r2[6];
            const int iy0 = y - 2, iy1 = y - 1, iy2 = y;
            const bool y0in = (iy0 >= 0) & (iy0 < s);
            const bool y1in = (iy1 >= 0) & (iy1 < s);
            const bool y2in = (iy2 >= 0) & (iy2 < s);
#pragma unroll
            for (int m = 0; m < 6; ++m) {
                int ix = x0 - 2 + m;
                bool xin = (ix >= 0) & (ix < s);
                r0[m] = (xin & y0in) ? in[iy0 * s + ix] : 0.0f;
                r1[m] = (xin & y1in) ? in[iy1 * s + ix] : 0.0f;
                r2[m] = (xin & y2in) ? in[iy2 * s + ix] : 0.0f;
            }
#pragma unroll
            for (int j = 0; j < 4; ++j) {
                int x = x0 + j;
                if (x < o) {
                    float acc = kb9;
                    acc += k0 * r0[j] + k1 * r0[j + 1] + k2 * r0[j + 2];
                    acc += k3 * r1[j] + k4 * r1[j + 1] + k5 * r1[j + 2];
                    acc += k6 * r2[j] + k7 * r2[j + 1] + k8 * r2[j + 2];
                    out[y * o + x] = acc;
                }
            }
        }
        s = o;
    }
}

// ---------------------------------------------------------------------------
// Phase C: pack W (256x256 fp32, final = wB) into bf16 hi/lo in the exact
// 16-bit B-matrix (32x16) WMMA fragment layout:
//   element (k,n) of K-tile kk, col-tile ct -> lane L = (n%16) + 16*((k%16)/8),
//   half e = (k%8) + 8*((k%32)/16).  Flat index: (kk*16+ct)*512 + L*16 + e.
// W uses the RNE split (one-time cost, best accuracy).
// Also extracts bias[256] = bm.flatten() (final = bB).
// ---------------------------------------------------------------------------
__global__ void pack_kernel(const float* __restrict__ Wsrc, const float* __restrict__ bsrc,
                            unsigned short* __restrict__ whi, unsigned short* __restrict__ wlo,
                            float* __restrict__ bias)
{
    if (blockIdx.x == 256) {
        int c = threadIdx.x;
        if (c < 256) bias[c] = bsrc[c];
        return;
    }
    int i   = blockIdx.x * 256 + threadIdx.x;   // 0..65535
    int kk  = i >> 13;
    int rem = i & 8191;
    int ct  = rem >> 9;
    int j   = rem & 511;
    int L   = j >> 4;
    int e   = j & 15;
    int k   = kk * 32 + (e & 7) + 8 * (L >> 4) + 16 * (e >> 3);
    int c   = ct * 16 + (L & 15);
    float v = Wsrc[k * 256 + c];
    unsigned short h = f2bf_rne(v);
    whi[i] = h;
    wlo[i] = f2bf_rne(v - bf2f(h));
}

// ---------------------------------------------------------------------------
// Phase D: GEMM  out[131072,256] = x @ W + b   (split-bf16, fp32 accumulate)
// 8 waves/block arranged 2(M) x 4(N); wave tile 32x64 (2x4 of 16x16);
// block tile 64 x 256 (full N -> x read exactly once from HBM).
// Per K=32 slice per wave: 2 A-frag conversions amortized over 4 N-tiles,
// 24 WMMAs (hi*hi + hi*lo + lo*hi per acc tile)  => ~5 conversion VALU ops
// per WMMA, under the VALU/WMMA co-execution budget.
// ---------------------------------------------------------------------------
__global__ __launch_bounds__(256)
void gemm_kernel(const float* __restrict__ x,
                 const unsigned short* __restrict__ whi,
                 const unsigned short* __restrict__ wlo,
                 const float* __restrict__ bias,
                 float* __restrict__ out)
{
    const int w  = threadIdx.x >> 5;       // wave 0..7
    const int L  = threadIdx.x & 31;       // lane
    const int lr = L & 15;
    const int lh = L >> 4;
    const int row0 = blockIdx.x * 64 + (w >> 2) * 32;   // 2 waves in M
    const int col0 = (w & 3) * 64;                      // 4 waves in N

    v8f acc[2][4] = {};

#pragma unroll
    for (int kk = 0; kk < 8; ++kk) {
        const int c0 = kk * 32 + 8 * lh;   // A-frag: halves 0-7 @c0, 8-15 @c0+16

        v16bf ahi[2], alo[2];
#pragma unroll
        for (int mi = 0; mi < 2; ++mi) {
            const float* p = x + (size_t)(row0 + mi * 16 + lr) * 256 + c0;
            float4 f0 = *(const float4*)(p);
            float4 f1 = *(const float4*)(p + 4);
            float4 f2 = *(const float4*)(p + 16);
            float4 f3 = *(const float4*)(p + 20);
            float tv[16] = { f0.x, f0.y, f0.z, f0.w, f1.x, f1.y, f1.z, f1.w,
                             f2.x, f2.y, f2.z, f2.w, f3.x, f3.y, f3.z, f3.w };
            u16x16 uh, ul;
#pragma unroll
            for (int e = 0; e < 16; ++e) {
                unsigned short hb, lb;
                split_trunc(tv[e], hb, lb);
                uh[e] = hb;
                ul[e] = lb;
            }
            ahi[mi] = __builtin_bit_cast(v16bf, uh);
            alo[mi] = __builtin_bit_cast(v16bf, ul);
        }

        v16bf bhi[4], blo[4];
#pragma unroll
        for (int ni = 0; ni < 4; ++ni) {
            const int ct  = (col0 >> 4) + ni;
            const int idx = (kk * 16 + ct) * 512 + L * 16;
            v8u h8 = *(const v8u*)(whi + idx);
            v8u l8 = *(const v8u*)(wlo + idx);
            bhi[ni] = __builtin_bit_cast(v16bf, h8);
            blo[ni] = __builtin_bit_cast(v16bf, l8);
        }

#pragma unroll
        for (int mi = 0; mi < 2; ++mi)
#pragma unroll
            for (int ni = 0; ni < 4; ++ni) {
                acc[mi][ni] = __builtin_amdgcn_wmma_f32_16x16x32_bf16(
                    false, ahi[mi], false, bhi[ni], (short)0, acc[mi][ni], false, false);
                acc[mi][ni] = __builtin_amdgcn_wmma_f32_16x16x32_bf16(
                    false, ahi[mi], false, blo[ni], (short)0, acc[mi][ni], false, false);
                acc[mi][ni] = __builtin_amdgcn_wmma_f32_16x16x32_bf16(
                    false, alo[mi], false, bhi[ni], (short)0, acc[mi][ni], false, false);
            }
    }

    // Epilogue: C/D layout -> element (vgpr v, lane L) = row v+8*(L/16), col L%16
#pragma unroll
    for (int ni = 0; ni < 4; ++ni) {
        const int cn = col0 + ni * 16 + lr;
        const float bv = bias[cn];
#pragma unroll
        for (int mi = 0; mi < 2; ++mi) {
            const int rbase = row0 + mi * 16 + 8 * lh;
#pragma unroll
            for (int v = 0; v < 8; ++v) {
                out[(size_t)(rbase + v) * 256 + cn] = acc[mi][ni][v] + bv;
            }
        }
    }
}

// ---------------------------------------------------------------------------
extern "C" void kernel_launch(void* const* d_in, const int* in_sizes, int n_in,
                              void* d_out, int out_size, void* d_ws, size_t ws_size,
                              hipStream_t stream)
{
    const float* x     = (const float*)d_in[0];
    const float* winit = (const float*)d_in[1];
    const float* binit = (const float*)d_in[2];
    const float* wW1   = (const float*)d_in[3];
    const float* wb1   = (const float*)d_in[4];
    const float* wW2   = (const float*)d_in[5];
    const float* wb2   = (const float*)d_in[6];
    const float* bW1   = (const float*)d_in[7];
    const float* bb1   = (const float*)d_in[8];
    const float* bW2   = (const float*)d_in[9];
    const float* bb2   = (const float*)d_in[10];

    char* ws = (char*)d_ws;
    float*          wA   = (float*)(ws);                       // 266240 B (<=254^2 fp32)
    float*          wB   = (float*)(ws + 266240);              // 266240 B (256^2 fp32)
    float*          bA   = (float*)(ws + 532480);              // 4096 B
    float*          bB   = (float*)(ws + 536576);              // 4096 B
    float*          wk   = (float*)(ws + 540672);              // 8192 B  (123*10 fp32)
    float*          bk   = (float*)(ws + 548864);              // 4096 B  (3*10 fp32)
    unsigned short* whi  = (unsigned short*)(ws + 552960);     // 131072 B
    unsigned short* wlo  = (unsigned short*)(ws + 684032);     // 131072 B
    float*          bias = (float*)(ws + 815104);              // 1024 B

    hyper_kernel<<<1, 64, 0, stream>>>(wW1, wb1, wW2, wb2, bW1, bb1, bW2, bb2, wk, bk);
    conv_rec_kernel<<<2, 1024, 0, stream>>>(winit, wk, wA, wB, binit, bk, bA, bB);
    // 123 steps: final weight matrix lands in wB; 3 steps: final bias matrix in bB.
    pack_kernel<<<257, 256, 0, stream>>>(wB, bB, whi, wlo, bias);
    gemm_kernel<<<2048, 256, 0, stream>>>(x, whi, wlo, bias, (float*)d_out);
}
